// Pointnet2StructurePointNet_6476810682601
// MI455X (gfx1250) — matrix operand order repaired and checked by reading, not compiled, not run on hardware
//
#include <hip/hip_runtime.h>

// ---------------------------------------------------------------------------
// PointNet++ (2x SA-MSG + prob head) for MI455X / gfx1250, wave32 + WMMA.
// All channel MLPs run on v_wmma_f32_16x16x32_f16 (f16 in, f32 accum),
// BN folded into per-channel alpha/beta epilogue, groups fused per-(b,s)
// workgroup so intermediates stay in LDS.
// ---------------------------------------------------------------------------

#define EPS_F 1e-5f
#define BB    16
#define NN    8192
#define NS1   128
#define NS2   64
#define CFEAT 224   // 32 + 64 + 128

typedef __attribute__((ext_vector_type(16))) _Float16 v16h;
typedef __attribute__((ext_vector_type(8)))  _Float16 v8h;
typedef __attribute__((ext_vector_type(8)))  float    v8f;

union Frag { v16h v; v8h q[2]; };

struct ConvBN { const float* W; const float* b; const float* g; const float* bt; };

__host__ __device__ constexpr int cmax(int a, int b) { return a > b ? a : b; }
__host__ __device__ constexpr int pad32(int c)       { return (c + 31) & ~31; }

// ---------------------------------------------------------------------------
// Wave-level GEMM tile engine: Y[ROWS x COUT] = relu?(alpha*(X W^T) + beta)
// X: ROWS x strideX (f16, LDS), W: COUT x CINP (f16, LDS, row = out channel).
// One 16x16 D-tile per WMMA; 4 waves (128-thread block) split the tiles.
// Fragment layouts follow CDNA5 ISA 7.12.2 (wave32).
// ---------------------------------------------------------------------------
template<int ROWS, int CINP, int COUT, bool RELU>
__device__ __forceinline__ void wave_gemm(const _Float16* X, int strideX,
                                          _Float16* Y, int strideY,
                                          const _Float16* W,
                                          const float* alpha, const float* beta)
{
    static_assert(ROWS % 16 == 0 && COUT % 16 == 0 && CINP % 32 == 0, "tile dims");
    const int lane = threadIdx.x & 31;
    const int wv   = threadIdx.x >> 5;
    const int n    = lane & 15;
    const int hf   = lane >> 4;
    constexpr int RT = ROWS / 16;
    constexpr int OT = COUT / 16;
    constexpr int TILES = RT * OT;
    for (int t = wv; t < TILES; t += 4) {
        const int rt = t / OT;
        const int ot = t - rt * OT;
        v8f acc = {};
        // A: lane holds row m = rt*16 + n; K cols 8*hf..+7 and 16+8*hf..+7 per chunk
        const _Float16* arow = X + (rt * 16 + n) * strideX + 8 * hf;
        // B: lane holds col n (out channel); K rows 8*hf..+7 and 16+8*hf..+7 per chunk
        const _Float16* brow = W + (ot * 16 + n) * CINP + 8 * hf;
#pragma unroll
        for (int kc = 0; kc < CINP; kc += 32) {
            Frag a, b;
            a.q[0] = *reinterpret_cast<const v8h*>(arow + kc);
            a.q[1] = *reinterpret_cast<const v8h*>(arow + kc + 16);
            b.q[0] = *reinterpret_cast<const v8h*>(brow + kc);
            b.q[1] = *reinterpret_cast<const v8h*>(brow + kc + 16);
            acc = __builtin_amdgcn_wmma_f32_16x16x32_f16(
                false, a.v, false, b.v, (short)0, acc, false, false);
        }
        const int o  = ot * 16 + n;
        const float al = alpha[o];
        const float be = beta[o];
#pragma unroll
        for (int r = 0; r < 8; ++r) {      // C/D: VGPR r -> row m = r + 8*hf
            float v = al * acc[r] + be;
            if (RELU) v = fmaxf(v, 0.0f);
            Y[(rt * 16 + r + 8 * hf) * strideY + o] = (_Float16)v;
        }
    }
}

__device__ __forceinline__ void zero_lds(_Float16* p, int n)
{
    for (int i = threadIdx.x; i < n; i += blockDim.x) p[i] = (_Float16)0.0f;
}

// Convert layer weights f32->f16 into LDS (zero-padded cin), fold BN.
__device__ __forceinline__ void load_weights(ConvBN L, int cout, int cinA, int cinP,
                                             _Float16* wbuf, float* alpha, float* beta)
{
    for (int i = threadIdx.x; i < cout * cinP; i += blockDim.x) {
        const int o = i / cinP;
        const int c = i - o * cinP;
        wbuf[i] = (c < cinA) ? (_Float16)L.W[o * cinA + c] : (_Float16)0.0f;
    }
    const float inv = rsqrtf(1.0f + EPS_F);
    for (int o = threadIdx.x; o < cout; o += blockDim.x) {
        const float a = L.g[o] * inv;
        alpha[o] = a;
        beta[o]  = a * L.b[o] + L.bt[o];
    }
}

// ---------------------------------------------------------------------------
// Farthest point sampling: one block per batch. LDS distance array + argmax.
// Matches jnp scan: idx[0]=0, idx[t+1]=argmax(min-dist after update t).
// ---------------------------------------------------------------------------
template<int NPTS, int NPOINT, int THREADS>
__global__ __launch_bounds__(THREADS)
void fps_kernel(const float* __restrict__ xyz, float* __restrict__ newxyz)
{
    __shared__ float dist[NPTS];
    __shared__ float sval[THREADS];
    __shared__ int   sidx[THREADS];
    __shared__ int   idxlist[NPOINT];
    __shared__ float fpt[3];
    __shared__ int   sfar;

    const int b = blockIdx.x, tid = threadIdx.x;
    const float* X = xyz + (size_t)b * NPTS * 3;

    for (int i = tid; i < NPTS; i += THREADS) dist[i] = 1e10f;
    if (tid == 0) sfar = 0;
    __syncthreads();

    for (int t = 0; t < NPOINT; ++t) {
        if (tid == 0) {
            const int far = sfar;
            idxlist[t] = far;
            fpt[0] = X[far * 3 + 0];
            fpt[1] = X[far * 3 + 1];
            fpt[2] = X[far * 3 + 2];
        }
        __syncthreads();
        const float fx = fpt[0], fy = fpt[1], fz = fpt[2];
        float best = -1.0f; int bi = 0;
        for (int i = tid; i < NPTS; i += THREADS) {
            const float dx = X[i * 3 + 0] - fx;
            const float dy = X[i * 3 + 1] - fy;
            const float dz = X[i * 3 + 2] - fz;
            const float d  = dx * dx + dy * dy + dz * dz;
            const float nd = fminf(dist[i], d);
            dist[i] = nd;
            if (nd > best) { best = nd; bi = i; }
        }
        sval[tid] = best; sidx[tid] = bi;
        __syncthreads();
        for (int off = THREADS / 2; off > 0; off >>= 1) {
            if (tid < off) {
                const float v = sval[tid + off];
                if (v > sval[tid] || (v == sval[tid] && sidx[tid + off] < sidx[tid])) {
                    sval[tid] = v; sidx[tid] = sidx[tid + off];
                }
            }
            __syncthreads();
        }
        if (tid == 0) sfar = sidx[0];
        __syncthreads();
    }
    for (int i = tid; i < NPOINT; i += THREADS) {
        const int id = idxlist[i];
        newxyz[((size_t)b * NPOINT + i) * 3 + 0] = X[id * 3 + 0];
        newxyz[((size_t)b * NPOINT + i) * 3 + 1] = X[id * 3 + 1];
        newxyz[((size_t)b * NPOINT + i) * 3 + 2] = X[id * 3 + 2];
    }
}

// ---------------------------------------------------------------------------
// Ball query: one wave per (b,s); first-K-by-index semantics via ballot/popc
// ordered compaction; remainder filled with first in-radius index.
// ---------------------------------------------------------------------------
__global__ __launch_bounds__(256)
void ball_query_kernel(const float* __restrict__ xyz, int npts,
                       const float* __restrict__ newxyz, int S,
                       int* __restrict__ out, int K, float r2, int npairs)
{
    const int lane = threadIdx.x & 31;
    const int p = blockIdx.x * 8 + (threadIdx.x >> 5);
    if (p >= npairs) return;
    const int b = p / S, s = p - b * S;
    const float* X = xyz + (size_t)b * npts * 3;
    const float cx = newxyz[((size_t)b * S + s) * 3 + 0];
    const float cy = newxyz[((size_t)b * S + s) * 3 + 1];
    const float cz = newxyz[((size_t)b * S + s) * 3 + 2];
    int* o = out + ((size_t)b * S + s) * K;

    int cnt = 0;
    int firstIdx = 0;
    bool haveFirst = false;
    for (int base = 0; base < npts && cnt < K; base += 32) {
        const int i = base + lane;
        bool inside = false;
        if (i < npts) {
            const float dx = X[i * 3 + 0] - cx;
            const float dy = X[i * 3 + 1] - cy;
            const float dz = X[i * 3 + 2] - cz;
            inside = (dx * dx + dy * dy + dz * dz) <= r2;
        }
        const unsigned mask = (unsigned)__ballot(inside);
        if (!haveFirst && mask) { firstIdx = base + __ffs(mask) - 1; haveFirst = true; }
        if (inside) {
            const int pos = cnt + __popc(mask & ((1u << lane) - 1u));
            if (pos < K) o[pos] = i;
        }
        cnt += __popc(mask);
    }
    if (cnt > K) cnt = K;
    for (int t = cnt + lane; t < K; t += 32) o[t] = firstIdx;
}

// ---------------------------------------------------------------------------
// Fused SA scale: gather group -> 3 WMMA conv-bn-relu layers -> max over K.
// One 128-thread workgroup per (b, s). Everything staged in dynamic LDS.
// ---------------------------------------------------------------------------
template<int K, int CIN, int C1, int C2, int C3>
struct SATraits {
    static constexpr int CP0  = pad32(CIN);
    static constexpr int CP1  = pad32(C1);
    static constexpr int CP2  = pad32(C2);
    static constexpr int CP3  = pad32(C3);
    static constexpr int BUF0 = K * cmax(CP0, CP2);
    static constexpr int BUF1 = K * cmax(CP1, CP3);
    static constexpr int WMAX = cmax(C1 * CP0, cmax(C2 * CP1, C3 * CP2));
    static constexpr size_t LDS = (size_t)(BUF0 + BUF1 + WMAX) * 2 + 2 * 128 * sizeof(float);
};

template<int K, int CIN, int C1, int C2, int C3, bool WITHFEATS>
__global__ __launch_bounds__(128)
void sa_scale_kernel(const float* __restrict__ xyz, int npts,
                     const float* __restrict__ newxyz, int S,
                     const int* __restrict__ ballidx,
                     const float* __restrict__ feats,  // (B, npts, CFEAT) or null
                     ConvBN L1, ConvBN L2, ConvBN L3,
                     float* __restrict__ featsOut, int chanOff, int totalChan)
{
    using T = SATraits<K, CIN, C1, C2, C3>;
    extern __shared__ __align__(16) char smem[];
    _Float16* buf0  = (_Float16*)smem;
    _Float16* buf1  = buf0 + T::BUF0;
    _Float16* wbuf  = buf1 + T::BUF1;
    float*    alpha = (float*)(wbuf + T::WMAX);
    float*    beta  = alpha + 128;

    const int tid = threadIdx.x;
    const int b = blockIdx.x / S;
    const int s = blockIdx.x - b * S;
    const int* bidx = ballidx + ((size_t)b * S + s) * K;
    const float cx = newxyz[((size_t)b * S + s) * 3 + 0];
    const float cy = newxyz[((size_t)b * S + s) * 3 + 1];
    const float cz = newxyz[((size_t)b * S + s) * 3 + 2];

    // Gather grouped input: [feats(id), xyz(id)-center], zero-padded to CP0.
    for (int i = tid; i < K * T::CP0; i += 128) {
        const int k = i / T::CP0;
        const int c = i - k * T::CP0;
        const int id = bidx[k];
        float v = 0.0f;
        if constexpr (WITHFEATS) {
            if (c < CFEAT) {
                v = feats[((size_t)b * npts + id) * CFEAT + c];
            } else if (c < CFEAT + 3) {
                const int d = c - CFEAT;
                v = xyz[((size_t)b * npts + id) * 3 + d] -
                    ((d == 0) ? cx : (d == 1) ? cy : cz);
            }
        } else {
            if (c < 3)
                v = xyz[((size_t)b * npts + id) * 3 + c] -
                    ((c == 0) ? cx : (c == 1) ? cy : cz);
        }
        buf0[i] = (_Float16)v;
    }
    zero_lds(buf1, K * T::CP1);
    load_weights(L1, C1, CIN, T::CP0, wbuf, alpha, beta);
    __syncthreads();
    wave_gemm<K, T::CP0, C1, true>(buf0, T::CP0, buf1, T::CP1, wbuf, alpha, beta);
    __syncthreads();

    zero_lds(buf0, K * T::CP2);
    load_weights(L2, C2, C1, T::CP1, wbuf, alpha, beta);
    __syncthreads();
    wave_gemm<K, T::CP1, C2, true>(buf1, T::CP1, buf0, T::CP2, wbuf, alpha, beta);
    __syncthreads();

    zero_lds(buf1, K * T::CP3);
    load_weights(L3, C3, C2, T::CP2, wbuf, alpha, beta);
    __syncthreads();
    wave_gemm<K, T::CP2, C3, true>(buf0, T::CP2, buf1, T::CP3, wbuf, alpha, beta);
    __syncthreads();

    // Max-pool over the K samples; output point-major (B, S, totalChan).
    for (int c = tid; c < C3; c += 128) {
        float m = 0.0f;  // post-ReLU values are >= 0
        for (int k = 0; k < K; ++k) m = fmaxf(m, (float)buf1[k * T::CP3 + c]);
        featsOut[((size_t)b * S + s) * totalChan + chanOff + c] = m;
    }
}

// ---------------------------------------------------------------------------
// Prob head: 5 conv-bn layers (relu on first 4) + softmax over 64 points +
// weighted xyz. One workgroup per batch.
// ---------------------------------------------------------------------------
struct ProbTraits {
    static constexpr int BUF0 = NS2 * 224;
    static constexpr int BUF1 = NS2 * 128;
    static constexpr int WMAX = 128 * 224;
    static constexpr size_t LDS = (size_t)(BUF0 + BUF1 + WMAX) * 2 + 2 * 128 * sizeof(float);
};

__global__ __launch_bounds__(128)
void prob_kernel(const float* __restrict__ feats2,   // (B, NS2, CFEAT)
                 const float* __restrict__ nxyz,     // (B, NS2, 3)
                 ConvBN P0, ConvBN P1, ConvBN P2, ConvBN P3, ConvBN P4,
                 float* __restrict__ out)            // (B, 16, 3)
{
    extern __shared__ __align__(16) char smem[];
    _Float16* buf0  = (_Float16*)smem;
    _Float16* buf1  = buf0 + ProbTraits::BUF0;
    _Float16* wbuf  = buf1 + ProbTraits::BUF1;
    float*    alpha = (float*)(wbuf + ProbTraits::WMAX);
    float*    beta  = alpha + 128;
    const int b = blockIdx.x, tid = threadIdx.x;

    for (int i = tid; i < NS2 * CFEAT; i += 128)
        buf0[i] = (_Float16)feats2[(size_t)b * NS2 * CFEAT + i];
    zero_lds(buf1, NS2 * 128);
    load_weights(P0, 128, 224, 224, wbuf, alpha, beta);
    __syncthreads();
    wave_gemm<NS2, 224, 128, true>(buf0, 224, buf1, 128, wbuf, alpha, beta);
    __syncthreads();

    zero_lds(buf0, NS2 * 64);
    load_weights(P1, 64, 128, 128, wbuf, alpha, beta);
    __syncthreads();
    wave_gemm<NS2, 128, 64, true>(buf1, 128, buf0, 64, wbuf, alpha, beta);
    __syncthreads();

    zero_lds(buf1, NS2 * 32);
    load_weights(P2, 32, 64, 64, wbuf, alpha, beta);
    __syncthreads();
    wave_gemm<NS2, 64, 32, true>(buf0, 64, buf1, 32, wbuf, alpha, beta);
    __syncthreads();

    zero_lds(buf0, NS2 * 32);
    load_weights(P3, 16, 32, 32, wbuf, alpha, beta);
    __syncthreads();
    wave_gemm<NS2, 32, 16, true>(buf1, 32, buf0, 32, wbuf, alpha, beta);
    __syncthreads();

    zero_lds(buf1, NS2 * 32);
    load_weights(P4, 16, 16, 32, wbuf, alpha, beta);   // cin 16 zero-padded to 32
    __syncthreads();
    wave_gemm<NS2, 32, 16, false>(buf0, 32, buf1, 32, wbuf, alpha, beta);
    __syncthreads();

    if (tid < 16) {
        const int m = tid;
        float mx = -1e30f;
        for (int n = 0; n < NS2; ++n) mx = fmaxf(mx, (float)buf1[n * 32 + m]);
        float sum = 0.0f, a0 = 0.0f, a1 = 0.0f, a2 = 0.0f;
        for (int n = 0; n < NS2; ++n) {
            const float p = __expf((float)buf1[n * 32 + m] - mx);
            sum += p;
            a0 += p * nxyz[((size_t)b * NS2 + n) * 3 + 0];
            a1 += p * nxyz[((size_t)b * NS2 + n) * 3 + 1];
            a2 += p * nxyz[((size_t)b * NS2 + n) * 3 + 2];
        }
        const float inv = 1.0f / sum;
        out[((size_t)b * 16 + m) * 3 + 0] = a0 * inv;
        out[((size_t)b * 16 + m) * 3 + 1] = a1 * inv;
        out[((size_t)b * 16 + m) * 3 + 2] = a2 * inv;
    }
}

// ---------------------------------------------------------------------------
// Host driver
// ---------------------------------------------------------------------------
extern "C" void kernel_launch(void* const* d_in, const int* in_sizes, int n_in,
                              void* d_out, int out_size, void* d_ws, size_t ws_size,
                              hipStream_t stream)
{
    (void)in_sizes; (void)n_in; (void)out_size; (void)ws_size;

    // d_in order: input, on_data, sa1 (3 scales x 3 layers x {W,b,g,bt}),
    //             sa2 (same), prob (5 layers x {W,b,g,bt})
    const float* onData = (const float*)d_in[1];
    const float* sa1p[3][3][4];
    const float* sa2p[3][3][4];
    const float* prp[5][4];
    for (int s = 0; s < 3; ++s)
        for (int l = 0; l < 3; ++l)
            for (int a = 0; a < 4; ++a) {
                sa1p[s][l][a] = (const float*)d_in[2  + (s * 3 + l) * 4 + a];
                sa2p[s][l][a] = (const float*)d_in[38 + (s * 3 + l) * 4 + a];
            }
    for (int l = 0; l < 5; ++l)
        for (int a = 0; a < 4; ++a)
            prp[l][a] = (const float*)d_in[74 + l * 4 + a];

    auto mk = [](const float* const p[4]) { return ConvBN{p[0], p[1], p[2], p[3]}; };

    // Workspace layout (~4.9 MB, 256B aligned chunks)
    size_t off = 0;
    auto alloc = [&](size_t bytes) -> void* {
        void* p = (char*)d_ws + off;
        off += (bytes + 255) & ~(size_t)255;
        return p;
    };
    float* newxyz1 = (float*)alloc((size_t)BB * NS1 * 3 * 4);
    int*   ball1a  = (int*)  alloc((size_t)BB * NS1 * 16 * 4);
    int*   ball1b  = (int*)  alloc((size_t)BB * NS1 * 32 * 4);
    int*   ball1c  = (int*)  alloc((size_t)BB * NS1 * 128 * 4);
    float* feats1  = (float*)alloc((size_t)BB * NS1 * CFEAT * 4);
    float* newxyz2 = (float*)alloc((size_t)BB * NS2 * 3 * 4);
    int*   ball2a  = (int*)  alloc((size_t)BB * NS2 * 16 * 4);
    int*   ball2b  = (int*)  alloc((size_t)BB * NS2 * 32 * 4);
    int*   ball2c  = (int*)  alloc((size_t)BB * NS2 * 128 * 4);
    float* feats2  = (float*)alloc((size_t)BB * NS2 * CFEAT * 4);

    // ---- Stage 1: FPS + ball query + fused MSG MLPs ----
    fps_kernel<NN, NS1, 1024><<<BB, 1024, 0, stream>>>(onData, newxyz1);

    ball_query_kernel<<<BB * NS1 / 8, 256, 0, stream>>>(onData, NN, newxyz1, NS1, ball1a, 16,  0.01f, BB * NS1);
    ball_query_kernel<<<BB * NS1 / 8, 256, 0, stream>>>(onData, NN, newxyz1, NS1, ball1b, 32,  0.04f, BB * NS1);
    ball_query_kernel<<<BB * NS1 / 8, 256, 0, stream>>>(onData, NN, newxyz1, NS1, ball1c, 128, 0.16f, BB * NS1);

    sa_scale_kernel<16, 3, 16, 16, 32, false>
        <<<BB * NS1, 128, SATraits<16, 3, 16, 16, 32>::LDS, stream>>>(
            onData, NN, newxyz1, NS1, ball1a, nullptr,
            mk(sa1p[0][0]), mk(sa1p[0][1]), mk(sa1p[0][2]), feats1, 0, CFEAT);
    sa_scale_kernel<32, 3, 32, 64, 64, false>
        <<<BB * NS1, 128, SATraits<32, 3, 32, 64, 64>::LDS, stream>>>(
            onData, NN, newxyz1, NS1, ball1b, nullptr,
            mk(sa1p[1][0]), mk(sa1p[1][1]), mk(sa1p[1][2]), feats1, 32, CFEAT);
    sa_scale_kernel<128, 3, 64, 96, 128, false>
        <<<BB * NS1, 128, SATraits<128, 3, 64, 96, 128>::LDS, stream>>>(
            onData, NN, newxyz1, NS1, ball1c, nullptr,
            mk(sa1p[2][0]), mk(sa1p[2][1]), mk(sa1p[2][2]), feats1, 96, CFEAT);

    // ---- Stage 2: FPS + ball query + fused MSG MLPs (with features) ----
    fps_kernel<NS1, NS2, 128><<<BB, 128, 0, stream>>>(newxyz1, newxyz2);

    ball_query_kernel<<<BB * NS2 / 8, 256, 0, stream>>>(newxyz1, NS1, newxyz2, NS2, ball2a, 16,  0.04f, BB * NS2);
    ball_query_kernel<<<BB * NS2 / 8, 256, 0, stream>>>(newxyz1, NS1, newxyz2, NS2, ball2b, 32,  0.16f, BB * NS2);
    ball_query_kernel<<<BB * NS2 / 8, 256, 0, stream>>>(newxyz1, NS1, newxyz2, NS2, ball2c, 128, 0.64f, BB * NS2);

    sa_scale_kernel<16, 227, 16, 16, 32, true>
        <<<BB * NS2, 128, SATraits<16, 227, 16, 16, 32>::LDS, stream>>>(
            newxyz1, NS1, newxyz2, NS2, ball2a, feats1,
            mk(sa2p[0][0]), mk(sa2p[0][1]), mk(sa2p[0][2]), feats2, 0, CFEAT);
    sa_scale_kernel<32, 227, 32, 64, 64, true>
        <<<BB * NS2, 128, SATraits<32, 227, 32, 64, 64>::LDS, stream>>>(
            newxyz1, NS1, newxyz2, NS2, ball2b, feats1,
            mk(sa2p[1][0]), mk(sa2p[1][1]), mk(sa2p[1][2]), feats2, 32, CFEAT);
    sa_scale_kernel<128, 227, 64, 96, 128, true>
        <<<BB * NS2, 128, SATraits<128, 227, 64, 96, 128>::LDS, stream>>>(
            newxyz1, NS1, newxyz2, NS2, ball2c, feats1,
            mk(sa2p[2][0]), mk(sa2p[2][1]), mk(sa2p[2][2]), feats2, 96, CFEAT);

    // ---- Prob head + softmax + weighted xyz ----
    prob_kernel<<<BB, 128, ProbTraits::LDS, stream>>>(
        feats2, newxyz2,
        mk(prp[0]), mk(prp[1]), mk(prp[2]), mk(prp[3]), mk(prp[4]),
        (float*)d_out);
}
